// SphereConv2DEXP_47820165873668
// MI455X (gfx1250) — matrix operand order repaired and checked by reading, not compiled, not run on hardware
//
#include <hip/hip_runtime.h>

typedef __attribute__((ext_vector_type(16))) __bf16 v16bf;
typedef __attribute__((ext_vector_type(8)))  float  v8f;
typedef __attribute__((ext_vector_type(4)))  unsigned int uint4v;

#define CIN   128
#define COUT  256
#define HH    128
#define WW    256
#define HWSZ  (HH * WW)          // 32768
#define KTOT  (CIN * 9)          // 1152
#define KC    32
#define NKC   (KTOT / KC)        // 36
#define PIX_TILE 32
#define NSITE (PIX_TILE * 9)     // 288
#define WT_PITCH 40              // bf16 per LDS weight row (32 + 8 pad -> 80B, conflict-free)
#define BT_PITCH 1160            // bf16 per LDS im2col row (1152 + 8 pad -> 2320B)

union V16 { v16bf bf; uint4v u4[2]; };

__device__ __forceinline__ unsigned short f2bf(float f) {
  union { float f; unsigned int u; } c; c.f = f;
  unsigned int u = c.u;
  u += 0x7FFFu + ((u >> 16) & 1u);        // round-to-nearest-even
  return (unsigned short)(u >> 16);
}

// ---- Kernel 1: repack weight f32 [co][k] -> bf16 [k/32][co][k%32] in workspace ----
__global__ void repack_weight_kernel(const float* __restrict__ w,
                                     unsigned short* __restrict__ wb) {
  int id = blockIdx.x * 256 + threadIdx.x;     // 0 .. 256*1152-1
  int co = id / KTOT;
  int k  = id - co * KTOT;
  int kc = k >> 5;
  int k32 = k & 31;
  wb[((size_t)(kc * COUT + co)) * KC + k32] = f2bf(w[id]);
}

struct __align__(16) Smem {
  unsigned short wt[2][COUT * WT_PITCH];     // 2 x 20480 B : double-buffered weight chunk
  unsigned short bt[PIX_TILE * BT_PITCH];    // 74240 B : im2col tile (bf16)
  int   sidx[NSITE][4];                      //  4608 B : bilinear corner indices
  float swt[NSITE][4];                       //  4608 B : bilinear corner weights
};                                           // total 124416 B (< 320 KB WGP LDS)

// ---- Kernel 2: fused grid-sample + im2col + WMMA GEMM + bias ----
__global__ void __launch_bounds__(256)
sphere_conv2d_wmma_kernel(const float* __restrict__ x,
                          const unsigned short* __restrict__ wb,
                          const float* __restrict__ bias,
                          const float* __restrict__ grid,
                          float* __restrict__ out) {
  __shared__ Smem sm;
  const int tid   = threadIdx.x;
  const int wgPix = blockIdx.x * PIX_TILE;         // 32 consecutive output pixels
  const int bIdx  = wgPix / HWSZ;                  // same batch for all 32 (HW % 32 == 0)
  const int sp0   = wgPix - bIdx * HWSZ;           // flattened (h,w) start
  const float* xb = x + (size_t)bIdx * CIN * HWSZ;

  // LDS byte addresses of this thread's async staging rows (row co = tid, both buffers)
  const unsigned ldsWt0 = (unsigned)(uintptr_t)&sm.wt[0][tid * WT_PITCH];
  const unsigned ldsWt1 = (unsigned)(uintptr_t)&sm.wt[1][tid * WT_PITCH];

  // Issue async copy of one 16 KB weight chunk: thread t copies row co=t (64 B).
  // async b128 copy: each lane moves 16 B from global memory into its LDS address
  auto stage = [&](int kc, unsigned ldsDst) {
    unsigned long long ga = (unsigned long long)(const void*)
        (wb + ((size_t)(kc * COUT + tid)) * KC);
    asm volatile(
        "global_load_async_to_lds_b128 %0, %1, off\n\t"
        "global_load_async_to_lds_b128 %0, %1, off offset:16\n\t"
        "global_load_async_to_lds_b128 %0, %1, off offset:32\n\t"
        "global_load_async_to_lds_b128 %0, %1, off offset:48"
        :: "v"(ldsDst), "v"(ga) : "memory");
  };

  // Kick off chunk 0 immediately: the copy overlaps the entire sampling phase.
  stage(0, ldsWt0);

  // ---------- Phase 1: per-site bilinear parameters (channel-invariant) ----------
  for (int s = tid; s < NSITE; s += 256) {
    int pixel = s / 9;
    int tap   = s - pixel * 9;
    int kh = tap / 3, kw = tap - kh * 3;
    int p    = wgPix + pixel;
    int wcol = p & (WW - 1);
    int hrow = (p >> 8) & (HH - 1);
    int gr = 3 * hrow + kh;
    int gc = 3 * wcol + kw;
    float gx = grid[((size_t)gr * (3 * WW) + gc) * 2 + 0];
    float gy = grid[((size_t)gr * (3 * WW) + gc) * 2 + 1];
    float ix = ((gx + 1.0f) * (float)WW - 1.0f) * 0.5f;
    float iy = ((gy + 1.0f) * (float)HH - 1.0f) * 0.5f;
    float x0f = floorf(ix), y0f = floorf(iy);
    float fx = ix - x0f, fy = iy - y0f;
    int x0 = (int)x0f, y0 = (int)y0f;
    int xs[2] = { x0, x0 + 1 };
    int ys[2] = { y0, y0 + 1 };
    float wxv[2] = { 1.0f - fx, fx };
    float wyv[2] = { 1.0f - fy, fy };
#pragma unroll
    for (int j = 0; j < 4; ++j) {
      int xc = xs[j & 1], yc = ys[j >> 1];
      bool valid = (xc >= 0) & (xc < WW) & (yc >= 0) & (yc < HH);
      int xi = xc < 0 ? 0 : (xc > WW - 1 ? WW - 1 : xc);
      int yi = yc < 0 ? 0 : (yc > HH - 1 ? HH - 1 : yc);
      sm.sidx[s][j] = yi * WW + xi;
      sm.swt[s][j]  = valid ? (wxv[j & 1] * wyv[j >> 1]) : 0.0f;
    }
  }
  __syncthreads();

  // ---------- Phase 2: build 32 x 1152 bf16 im2col tile in LDS ----------
#pragma unroll 1
  for (int it = 0; it < (PIX_TILE * CIN * 9) / 256; ++it) {   // 144 iters
    int g    = it * 256 + tid;
    int ci   = g / NSITE;
    int site = g - ci * NSITE;
    int pixel = site / 9;
    int tap   = site - pixel * 9;
    const int*   si = sm.sidx[site];
    const float* sw = sm.swt[site];
    const float* base = xb + ci * HWSZ;
    float v = sw[0] * base[si[0]] + sw[1] * base[si[1]]
            + sw[2] * base[si[2]] + sw[3] * base[si[3]];
    sm.bt[pixel * BT_PITCH + ci * 9 + tap] = f2bf(v);
  }

  // ---------- Phase 3: WMMA GEMM, double-buffered async weight staging ----------
  // M = cout (2 tiles/wave: wave, wave+8), N = pixel (2 tiles: 0-15, 16-31), K steps of 32.
  const int lane  = tid & 31;
  const int wave  = tid >> 5;
  const int laneM = lane & 15;
  const int hi    = lane >> 4;                 // 0: lanes 0-15, 1: lanes 16-31
  const int cobase0 = wave * 16;
  const int cobase1 = cobase0 + 128;
  // A (16-bit 16x32) per-lane K chunks: lo-half {0..7,16..23}, hi-half {8..15,24..31}
  const int kA0 = hi ? 8 : 0;
  const int kA1 = hi ? 24 : 16;
  v8f acc00 = {}; v8f acc01 = {}; v8f acc10 = {}; v8f acc11 = {};

#pragma unroll 1
  for (int kc = 0; kc < NKC; ++kc) {
    const int p = kc & 1;
    // chunk kc (issued one iteration ago, or in prologue) has landed in wt[p]:
    asm volatile("s_wait_asynccnt 0x0" ::: "memory");
    __syncthreads();   // all waves' copies landed; prior reads of wt[1-p] retired
    if (kc + 1 < NKC)
      stage(kc + 1, p ? ldsWt0 : ldsWt1);      // overlap next copy with compute
    if (kc + 2 < NKC)
      __builtin_prefetch(wb + ((size_t)((kc + 2) * COUT + tid)) * KC, 0, 0);

    const unsigned short* wtp = sm.wt[p];
    V16 a0, a1, b0, b1;
    a0.u4[0] = *(const uint4v*)&wtp[(cobase0 + laneM) * WT_PITCH + kA0];
    a0.u4[1] = *(const uint4v*)&wtp[(cobase0 + laneM) * WT_PITCH + kA1];
    a1.u4[0] = *(const uint4v*)&wtp[(cobase1 + laneM) * WT_PITCH + kA0];
    a1.u4[1] = *(const uint4v*)&wtp[(cobase1 + laneM) * WT_PITCH + kA1];
    // B (32x16): lane holds column N=laneM, K chunks lo-half {0..7,16..23}, hi-half {8..15,24..31}
    const int kb = kc * KC + hi * 8;
    b0.u4[0] = *(const uint4v*)&sm.bt[laneM * BT_PITCH + kb];
    b0.u4[1] = *(const uint4v*)&sm.bt[laneM * BT_PITCH + kb + 16];
    b1.u4[0] = *(const uint4v*)&sm.bt[(laneM + 16) * BT_PITCH + kb];
    b1.u4[1] = *(const uint4v*)&sm.bt[(laneM + 16) * BT_PITCH + kb + 16];

    acc00 = __builtin_amdgcn_wmma_f32_16x16x32_bf16(false, a0.bf, false, b0.bf,
                                                    (short)0, acc00, false, false);
    acc01 = __builtin_amdgcn_wmma_f32_16x16x32_bf16(false, a0.bf, false, b1.bf,
                                                    (short)0, acc01, false, false);
    acc10 = __builtin_amdgcn_wmma_f32_16x16x32_bf16(false, a1.bf, false, b0.bf,
                                                    (short)0, acc10, false, false);
    acc11 = __builtin_amdgcn_wmma_f32_16x16x32_bf16(false, a1.bf, false, b1.bf,
                                                    (short)0, acc11, false, false);
  }

  // ---------- Epilogue: D layout -> NCHW, add bias ----------
  // VGPR v: lanes 0-15 -> M=v, lanes 16-31 -> M=v+8; N = laneM
  size_t outBase = (size_t)bIdx * COUT * HWSZ + sp0 + laneM;
#pragma unroll
  for (int v = 0; v < 8; ++v) {
    int co0 = cobase0 + v + hi * 8;
    int co1 = cobase1 + v + hi * 8;
    float b0f = bias[co0], b1f = bias[co1];
    out[outBase + (size_t)co0 * HWSZ]      = acc00[v] + b0f;
    out[outBase + (size_t)co0 * HWSZ + 16] = acc01[v] + b0f;
    out[outBase + (size_t)co1 * HWSZ]      = acc10[v] + b1f;
    out[outBase + (size_t)co1 * HWSZ + 16] = acc11[v] + b1f;
  }
}

extern "C" void kernel_launch(void* const* d_in, const int* in_sizes, int n_in,
                              void* d_out, int out_size, void* d_ws, size_t ws_size,
                              hipStream_t stream) {
  (void)in_sizes; (void)n_in; (void)out_size; (void)ws_size;
  const float* x      = (const float*)d_in[0];
  const float* weight = (const float*)d_in[1];
  const float* bias   = (const float*)d_in[2];
  const float* grid   = (const float*)d_in[3];
  unsigned short* wb  = (unsigned short*)d_ws;   // 256*1152 bf16 = 589,824 B

  repack_weight_kernel<<<(COUT * KTOT) / 256, 256, 0, stream>>>(weight, wb);

  const int nblocks = (4 * HWSZ) / PIX_TILE;     // 4096 workgroups
  sphere_conv2d_wmma_kernel<<<nblocks, 256, 0, stream>>>(
      x, wb, bias, grid, (float*)d_out);
}
